// MetaEdgePooling_56908316672612
// MI455X (gfx1250) — compile-verified, matrix operand order checked by reading, not run on hardware
//
#include <hip/hip_runtime.h>
#include <stdint.h>

#define DIM 256
#define ROUNDS 64
#define GPAD 8

typedef __attribute__((ext_vector_type(16))) _Float16 v16h;
typedef __attribute__((ext_vector_type(8)))  _Float16 v8h;
typedef __attribute__((ext_vector_type(4)))  _Float16 v4h;
typedef __attribute__((ext_vector_type(8)))  float    v8f;
typedef int v2i_g __attribute__((vector_size(2 * sizeof(int))));  // matches builtin param

#define AS1 __attribute__((address_space(1)))
#define AS3 __attribute__((address_space(3)))

__device__ __forceinline__ unsigned f2ord(float f) {
    unsigned u = __float_as_uint(f);
    return (u & 0x80000000u) ? ~u : (u | 0x80000000u);
}
__device__ __forceinline__ float ord2f(unsigned o) {
    unsigned u = (o & 0x80000000u) ? (o & 0x7FFFFFFFu) : ~o;
    return __uint_as_float(u);
}

// ---------------------------------------------------------------------------
// Stage 1: dual GEMV via WMMA, LDS-staged.
// Block = 128 threads (4 waves) = 64 rows of x. Cooperative coalesced float4
// loads -> f16 LDS tile; each wave runs 8 unrolled v_wmma_f32_16x16x32_f16.
// B fragment: col0 = W_src, col1 = W_dst, rest zero.
// ---------------------------------------------------------------------------
__global__ void __launch_bounds__(128)
k_gemv_wmma(const float* __restrict__ x, const float* __restrict__ W,
            float* __restrict__ s_src, float* __restrict__ s_dst, int N) {
    __shared__ __align__(16) _Float16 xs[64][DIM + GPAD];
    __shared__ __align__(16) _Float16 wk[2 * DIM];

    const int tid    = threadIdx.x;
    const int block0 = blockIdx.x * 64;
    int rowsHere = N - block0;
    if (rowsHere > 64) rowsHere = 64;

    for (int i = tid; i < 2 * DIM; i += 128) wk[i] = (_Float16)W[i];

    const float4* xg = (const float4*)(x + (size_t)block0 * DIM);
    const int totalVec = rowsHere * (DIM / 4);
    for (int i = tid; i < totalVec; i += 128) {
        float4 g = xg[i];
        int row = i >> 6;
        int c4  = (i & 63) << 2;
        v4h h4;
        h4[0] = (_Float16)g.x; h4[1] = (_Float16)g.y;
        h4[2] = (_Float16)g.z; h4[3] = (_Float16)g.w;
        *(v4h*)&xs[row][c4] = h4;
    }
    __syncthreads();

    const int wtile = tid >> 5;
    const int lane  = tid & 31;
    const int row   = lane & 15;   // A row within tile == B/D column index
    const int hi    = lane >> 4;   // K sub-block selector
    const int h     = hi * 8;
    const int tileRow0 = block0 + wtile * 16;
    if (tileRow0 >= N) return;

    const _Float16* xr  = &xs[wtile * 16 + row][0];
    const _Float16* wp  = &wk[((row < 2) ? row : 0) * DIM];
    const _Float16  msk = (row < 2) ? (_Float16)1.0f : (_Float16)0.0f;

    v8f c = {};
#pragma unroll
    for (int k = 0; k < DIM; k += 32) {
        v8h alo = *(const v8h*)(xr + k + h);
        v8h ahi = *(const v8h*)(xr + k + 16 + h);
        v8h blo = *(const v8h*)(wp + k + h);
        v8h bhi = *(const v8h*)(wp + k + 16 + h);
        blo *= msk;
        bhi *= msk;
        v16h a, bf;
#pragma unroll
        for (int j = 0; j < 8; ++j) {
            a[j]      = alo[j];
            a[8 + j]  = ahi[j];
            bf[j]     = blo[j];
            bf[8 + j] = bhi[j];
        }
        c = __builtin_amdgcn_wmma_f32_16x16x32_f16(false, a, false, bf,
                                                   (short)0, c, false, false);
    }
    // D layout: VGPR r holds (M=r, lanes0-15) / (M=r+8, lanes16-31), N = lane&15
    int rbase = tileRow0 + 8 * hi;
    if (row == 0) {
#pragma unroll
        for (int r = 0; r < 8; ++r) s_src[rbase + r] = c[r];
    } else if (row == 1) {
#pragma unroll
        for (int r = 0; r < 8; ++r) s_dst[rbase + r] = c[r];
    }
}

// ---------------------------------------------------------------------------
// Init
// ---------------------------------------------------------------------------
__global__ void k_init_nodes(unsigned* node_max, float* node_sum, unsigned* matched,
                             float* nes, int* mcnt, int* cluster, int* out_batch, int N) {
    int v = blockIdx.x * blockDim.x + threadIdx.x;
    if (v >= N) return;
    node_max[v] = 0u;
    node_sum[v] = 0.0f;
    matched[v]  = 0u;
    nes[v]      = 1.0f;
    mcnt[v]     = 0;
    cluster[v]  = 0;
    out_batch[v] = 0;
}
__global__ void k_init_edges(unsigned* alive, unsigned* chosen, int E) {
    int e = blockIdx.x * blockDim.x + threadIdx.x;
    if (e >= E) return;
    alive[e]  = 1u;
    chosen[e] = 0u;
}
__global__ void k_init_misc(int* cnt, int* nMatched, int E) {
    if (blockIdx.x == 0 && threadIdx.x == 0) {
        cnt[0] = E; cnt[1] = 0; nMatched[0] = 0;
    }
}

// ---------------------------------------------------------------------------
// Edge scores + segment softmax over dst
// ---------------------------------------------------------------------------
__global__ void k_edge_raw(const int* __restrict__ src, const int* __restrict__ dst,
                           const float* __restrict__ s_src, const float* __restrict__ s_dst,
                           const float* __restrict__ b, float* __restrict__ raw,
                           unsigned* node_max, int E) {
    int e = blockIdx.x * blockDim.x + threadIdx.x;
    if (e >= E) return;
    float r = s_src[src[e]] + s_dst[dst[e]] + b[0];
    raw[e] = r;
    atomicMax(&node_max[dst[e]], f2ord(r));
}
__global__ void k_edge_ex(const int* __restrict__ dst, float* __restrict__ raw,
                          const unsigned* __restrict__ node_max, float* node_sum, int E) {
    int e = blockIdx.x * blockDim.x + threadIdx.x;
    if (e >= E) return;
    float m  = ord2f(node_max[dst[e]]);
    float ex = __expf(raw[e] - m);
    raw[e] = ex;
    atomicAdd(&node_sum[dst[e]], ex);
}
__global__ void k_edge_score(const int* __restrict__ dst, const float* __restrict__ raw,
                             const float* __restrict__ node_sum, float* __restrict__ esc,
                             unsigned long long* __restrict__ key, int E) {
    int e = blockIdx.x * blockDim.x + threadIdx.x;
    if (e >= E) return;
    float v = raw[e] / node_sum[dst[e]] + 0.5f;   // ADD_TO_EDGE_SCORE
    esc[e] = v;
    // total order: ascending key == descending score, ties by ascending edge id
    key[e] = ((unsigned long long)(~f2ord(v)) << 32) | (unsigned)e;
}

// ---------------------------------------------------------------------------
// Locally-dominant greedy matching rounds (fixed count, device early-exit)
// ---------------------------------------------------------------------------
__global__ void k_round_reset(unsigned long long* node_min, int* cnt, int cur, int N) {
    if (blockIdx.x == 0 && threadIdx.x == 0) cnt[cur ^ 1] = 0;  // unconditional
    if (cnt[cur] == 0) return;
    int v = blockIdx.x * blockDim.x + threadIdx.x;
    if (v < N) node_min[v] = ~0ULL;
}
__global__ void k_round_min(const int* __restrict__ src, const int* __restrict__ dst,
                            const unsigned* __restrict__ alive,
                            const unsigned long long* __restrict__ key,
                            unsigned long long* node_min, const int* cnt, int cur, int E) {
    if (cnt[cur] == 0) return;
    int e = blockIdx.x * blockDim.x + threadIdx.x;
    if (e >= E) return;
    __builtin_prefetch(&key[(e + 4096 < E) ? e + 4096 : e], 0, 1);
    if (!alive[e]) return;
    unsigned long long k = key[e];
    atomicMin(&node_min[src[e]], k);
    atomicMin(&node_min[dst[e]], k);
}
__global__ void k_round_pick(const int* __restrict__ src, const int* __restrict__ dst,
                             const unsigned* __restrict__ alive,
                             const unsigned long long* __restrict__ key,
                             const unsigned long long* __restrict__ node_min,
                             unsigned* matched, unsigned* chosen,
                             const int* cnt, int cur, int E) {
    if (cnt[cur] == 0) return;
    int e = blockIdx.x * blockDim.x + threadIdx.x;
    if (e >= E || !alive[e]) return;
    unsigned long long k = key[e];
    if (k == node_min[src[e]] && k == node_min[dst[e]]) {
        chosen[e] = 1u;
        matched[src[e]] = 1u;   // picks are endpoint-disjoint (keys unique)
        matched[dst[e]] = 1u;
    }
}
__global__ void k_round_alive(const int* __restrict__ src, const int* __restrict__ dst,
                              unsigned* alive, const unsigned* __restrict__ matched,
                              int* cnt, int cur, int E) {
    if (cnt[cur] == 0) return;
    int e = blockIdx.x * blockDim.x + threadIdx.x;
    if (e >= E || !alive[e]) return;
    if (matched[src[e]] || matched[dst[e]]) { alive[e] = 0u; return; }
    atomicAdd(&cnt[cur ^ 1], 1);
}

// ---------------------------------------------------------------------------
// Cluster-id assignment
// ---------------------------------------------------------------------------
__global__ void k_compact(const unsigned* __restrict__ chosen, int* nMatched,
                          int* list, int E) {
    int e = blockIdx.x * blockDim.x + threadIdx.x;
    if (e >= E || !chosen[e]) return;
    int p = atomicAdd(nMatched, 1);
    list[p] = e;
}
// cid[i] = #chosen edges with smaller key (== position in score-sorted order)
__global__ void k_cid(const unsigned long long* __restrict__ key,
                      const int* __restrict__ list, const int* __restrict__ nMatched,
                      int* cid) {
    __shared__ unsigned long long tile[256];
    int nC = *nMatched;
    int i  = blockIdx.x * 256 + threadIdx.x;
    unsigned long long myk = (i < nC) ? key[list[i]] : ~0ULL;
    int cnt = 0;
    for (int j0 = 0; j0 < nC; j0 += 256) {
        int j  = j0 + (int)threadIdx.x;
        int jj = (j < nC) ? j : (nC - 1);   // clamp (fixed up below)
#if __has_builtin(__builtin_amdgcn_global_load_async_to_lds_b64)
        // gfx1250 async-to-LDS path: VMEM gather written straight into LDS
        __builtin_amdgcn_global_load_async_to_lds_b64(
            (AS1 v2i_g*)(v2i_g*)&key[list[jj]],
            (AS3 v2i_g*)(v2i_g*)&tile[threadIdx.x], 0, 0);
#if __has_builtin(__builtin_amdgcn_s_wait_asynccnt)
        __builtin_amdgcn_s_wait_asynccnt(0);
#else
        asm volatile("s_wait_asynccnt 0x0" ::: "memory");
#endif
        __syncthreads();
        if (j >= nC) tile[threadIdx.x] = ~0ULL;
#else
        tile[threadIdx.x] = (j < nC) ? key[list[jj]] : ~0ULL;
#endif
        __syncthreads();
        int lim = (nC - j0 < 256) ? (nC - j0) : 256;
        for (int t = 0; t < lim; ++t) cnt += (tile[t] < myk) ? 1 : 0;
        __syncthreads();
    }
    if (i < nC) cid[i] = cnt;
}
__global__ void k_scatter_chosen(const int* __restrict__ list, const int* __restrict__ cid,
                                 const int* __restrict__ nMatched,
                                 const int* __restrict__ src, const int* __restrict__ dst,
                                 const float* __restrict__ esc,
                                 int* cluster, float* nes) {
    int i = blockIdx.x * blockDim.x + threadIdx.x;
    if (i >= *nMatched) return;
    int e = list[i], c = cid[i];
    cluster[src[e]] = c;
    cluster[dst[e]] = c;
    nes[c] = esc[e];
}

// ---------------------------------------------------------------------------
// Prefix scan over unmatched nodes -> singleton cluster ids
// ---------------------------------------------------------------------------
__global__ void k_scan1(const unsigned* __restrict__ matched, int* blockSums, int N) {
    __shared__ int sdata[256];
    int v = blockIdx.x * 256 + threadIdx.x;
    sdata[threadIdx.x] = (v < N && !matched[v]) ? 1 : 0;
    __syncthreads();
    for (int s = 128; s > 0; s >>= 1) {
        if ((int)threadIdx.x < s) sdata[threadIdx.x] += sdata[threadIdx.x + s];
        __syncthreads();
    }
    if (threadIdx.x == 0) blockSums[blockIdx.x] = sdata[0];
}
__global__ void k_scan2(const int* blockSums, int* blockOff, int nBlocks,
                        const int* nMatched, int* out_num_clusters) {
    if (blockIdx.x == 0 && threadIdx.x == 0) {
        int run = 0;
        for (int b = 0; b < nBlocks; ++b) { blockOff[b] = run; run += blockSums[b]; }
        out_num_clusters[0] = *nMatched + run;
    }
}
__global__ void k_scan3(const unsigned* __restrict__ matched, const int* __restrict__ blockOff,
                        const int* __restrict__ nMatched, int* cluster, int N) {
    __shared__ int sdata[256];
    int v = blockIdx.x * 256 + threadIdx.x;
    int val = (v < N && !matched[v]) ? 1 : 0;
    sdata[threadIdx.x] = val;
    __syncthreads();
    for (int s = 1; s < 256; s <<= 1) {
        int t = ((int)threadIdx.x >= s) ? sdata[threadIdx.x - s] : 0;
        __syncthreads();
        sdata[threadIdx.x] += t;
        __syncthreads();
    }
    if (v < N && val) {
        int excl = sdata[threadIdx.x] - 1;  // exclusive prefix for this element
        cluster[v] = *nMatched + blockOff[blockIdx.x] + excl;
    }
}

// ---------------------------------------------------------------------------
// Merge features + remap edges/batch
// ---------------------------------------------------------------------------
__global__ void k_members(const int* __restrict__ cluster, int* mcnt, int* members, int N) {
    int v = blockIdx.x * blockDim.x + threadIdx.x;
    if (v >= N) return;
    int c = cluster[v];
    int slot = atomicAdd(&mcnt[c], 1);
    if (slot < 2) members[2 * c + slot] = v;
}
__global__ void k_newx(const float* __restrict__ x, const int* __restrict__ mcnt,
                       const int* __restrict__ members, const float* __restrict__ nes,
                       float* __restrict__ newx) {
    int c = blockIdx.x;
    int f = threadIdx.x;
    int n = mcnt[c]; if (n > 2) n = 2;
    float acc = 0.0f;
    for (int k = 0; k < n; ++k)
        acc += x[(size_t)members[2 * c + k] * DIM + f];
    newx[(size_t)c * DIM + f] = acc * nes[c];
}
__global__ void k_newei(const int* __restrict__ ei, const int* __restrict__ cluster,
                        int* __restrict__ out, int n2e) {
    int i = blockIdx.x * blockDim.x + threadIdx.x;
    if (i >= n2e) return;
    out[i] = cluster[ei[i]];
}
__global__ void k_newbatch(const int* __restrict__ batch, const int* __restrict__ cluster,
                           int* out_batch, int N) {
    int v = blockIdx.x * blockDim.x + threadIdx.x;
    if (v >= N) return;
    out_batch[cluster[v]] = batch[v];
}

// ---------------------------------------------------------------------------
extern "C" void kernel_launch(void* const* d_in, const int* in_sizes, int n_in,
                              void* d_out, int out_size, void* d_ws, size_t ws_size,
                              hipStream_t stream) {
    const float* x     = (const float*)d_in[0];
    const int*   ei    = (const int*)d_in[1];
    const int*   batch = (const int*)d_in[2];
    const float* W     = (const float*)d_in[3];
    const float* b     = (const float*)d_in[4];

    const int N  = in_sizes[2];          // 100000
    const int E  = in_sizes[1] / 2;      // 1600000
    const int*   src = ei;
    const int*   dst = ei + E;

    // d_out layout: new_x[N*DIM] f32 | new_edge_index[2E] i32 | new_batch[N] i32 | num_clusters i32
    float* out_x  = (float*)d_out;
    int*   out_ei = (int*)((float*)d_out + (size_t)N * DIM);
    int*   out_b  = out_ei + 2 * E;
    int*   out_nc = out_b + N;

    // workspace carve (256B aligned)
    char* p = (char*)d_ws;
    auto carve = [&](size_t bytes) {
        void* r = (void*)p;
        p += (bytes + 255) & ~(size_t)255;
        return r;
    };
    float*              s_src    = (float*)carve((size_t)N * 4);
    float*              s_dst    = (float*)carve((size_t)N * 4);
    float*              raw      = (float*)carve((size_t)E * 4);
    float*              esc      = (float*)carve((size_t)E * 4);
    unsigned long long* key      = (unsigned long long*)carve((size_t)E * 8);
    unsigned long long* node_min = (unsigned long long*)carve((size_t)N * 8);
    unsigned*           node_max = (unsigned*)carve((size_t)N * 4);
    float*              node_sum = (float*)carve((size_t)N * 4);
    unsigned*           alive    = (unsigned*)carve((size_t)E * 4);
    unsigned*           chosen   = (unsigned*)carve((size_t)E * 4);
    unsigned*           matched  = (unsigned*)carve((size_t)N * 4);
    int*                cluster  = (int*)carve((size_t)N * 4);
    float*              nes      = (float*)carve((size_t)N * 4);
    int*                mcnt     = (int*)carve((size_t)N * 4);
    int*                members  = (int*)carve((size_t)2 * N * 4);
    int*                list     = (int*)carve((size_t)N * 4);
    int*                cid      = (int*)carve((size_t)N * 4);
    int*                blockSums= (int*)carve(1024 * 4);
    int*                blockOff = (int*)carve(1024 * 4);
    int*                cnt      = (int*)carve(64);
    int*                nMatched = (int*)carve(64);

    const int TB = 256;
    const int nBlocksN = (N + TB - 1) / TB;
    const int nBlocksE = (E + TB - 1) / TB;
    const int n2E      = 2 * E;
    const int nBlocks2E = (n2E + TB - 1) / TB;

    // init
    k_init_nodes<<<nBlocksN, TB, 0, stream>>>(node_max, node_sum, matched, nes, mcnt,
                                              cluster, out_b, N);
    k_init_edges<<<nBlocksE, TB, 0, stream>>>(alive, chosen, E);
    k_init_misc<<<1, 64, 0, stream>>>(cnt, nMatched, E);

    // stage 1: WMMA dual-GEMV (64 rows per block, LDS-staged, 8 WMMAs/wave)
    k_gemv_wmma<<<(N + 63) / 64, 128, 0, stream>>>(x, W, s_src, s_dst, N);

    // stage 2: edge scores + segment softmax
    k_edge_raw<<<nBlocksE, TB, 0, stream>>>(src, dst, s_src, s_dst, b, raw, node_max, E);
    k_edge_ex<<<nBlocksE, TB, 0, stream>>>(dst, raw, node_max, node_sum, E);
    k_edge_score<<<nBlocksE, TB, 0, stream>>>(dst, raw, node_sum, esc, key, E);

    // stage 3: greedy matching rounds
    for (int r = 0; r < ROUNDS; ++r) {
        int cur = r & 1;
        k_round_reset<<<nBlocksN, TB, 0, stream>>>(node_min, cnt, cur, N);
        k_round_min<<<nBlocksE, TB, 0, stream>>>(src, dst, alive, key, node_min, cnt, cur, E);
        k_round_pick<<<nBlocksE, TB, 0, stream>>>(src, dst, alive, key, node_min,
                                                  matched, chosen, cnt, cur, E);
        k_round_alive<<<nBlocksE, TB, 0, stream>>>(src, dst, alive, matched, cnt, cur, E);
    }

    // stage 4: cluster ids
    k_compact<<<nBlocksE, TB, 0, stream>>>(chosen, nMatched, list, E);
    k_cid<<<nBlocksN, TB, 0, stream>>>(key, list, nMatched, cid);
    k_scatter_chosen<<<nBlocksN, TB, 0, stream>>>(list, cid, nMatched, src, dst, esc,
                                                  cluster, nes);
    k_scan1<<<nBlocksN, TB, 0, stream>>>(matched, blockSums, N);
    k_scan2<<<1, 64, 0, stream>>>(blockSums, blockOff, nBlocksN, nMatched, out_nc);
    k_scan3<<<nBlocksN, TB, 0, stream>>>(matched, blockOff, nMatched, cluster, N);

    // stage 5: merge + remap
    k_members<<<nBlocksN, TB, 0, stream>>>(cluster, mcnt, members, N);
    k_newx<<<N, DIM, 0, stream>>>(x, mcnt, members, nes, out_x);
    k_newei<<<nBlocks2E, TB, 0, stream>>>(ei, cluster, out_ei, n2E);
    k_newbatch<<<nBlocksN, TB, 0, stream>>>(batch, cluster, out_b, N);

    (void)n_in; (void)out_size; (void)ws_size;
}
